// DequantingLinear_52321291600113
// MI455X (gfx1250) — matrix operand order repaired and checked by reading, not compiled, not run on hardware
//
#include <hip/hip_runtime.h>

// y = x @ W^T + b with Q8_0-style blockwise dequant, mapped onto
// V_WMMA_F32_16X16X32_F16 (WMMA K == quant block == 32).
// M=64, N=12288, K=3072.
//
// Launch shape: 768 workgroups x 128 threads (4 waves). Each workgroup owns a
// 16-wide N tile; the 96 K-blocks are split 24-per-wave so 3072 waves are in
// flight (latency hiding for the HBM-bound weight stream), and per-wave 4
// fp32 accumulators cover all 64 M rows so each dequantized B fragment is
// reused 4x -> the 151 MB weight stream is read from HBM exactly once.
// Partial sums are combined through LDS with one workgroup barrier.

#define IN_F   3072
#define OUT_F  12288
#define NBLK   (IN_F / 32)    // 96 quant blocks along K
#define KSPLIT 4              // waves per workgroup (K-split)
#define BLK_PER_WAVE (NBLK / KSPLIT)   // 24

typedef __attribute__((ext_vector_type(16))) _Float16 v16h;
typedef __attribute__((ext_vector_type(8)))  float    v8f;
typedef __attribute__((ext_vector_type(4)))  float    v4f;
typedef __attribute__((ext_vector_type(4)))  int      v4i;

__global__ __launch_bounds__(32 * KSPLIT)
void DequantingLinear_52321291600113_kernel(const float* __restrict__ x,
                                            const int*   __restrict__ wq,
                                            const float* __restrict__ wsc,
                                            const int*   __restrict__ bq,
                                            const float* __restrict__ bsc,
                                            float* __restrict__ out)
{
    const int lane = threadIdx.x & 31;     // wave32 lane
    const int wv   = threadIdx.x >> 5;     // wave id 0..3 (K-split)
    const int nloc = lane & 15;            // N within tile (B/C), M within tile (A)
    const int half = lane >> 4;            // lane group 0/1
    const int ncol = blockIdx.x * 16 + nloc;

    v8f acc[4] = {};                       // 4 M-tiles x (16x16 f32)

    const int*   wrow = wq  + (size_t)ncol * (NBLK * 32) + half * 16;
    const float* srow = wsc + (size_t)ncol * NBLK;

    const int b0 = wv * BLK_PER_WAVE;
    const int b1 = b0 + BLK_PER_WAVE;

    for (int b = b0; b < b1; ++b) {
        // ---- B fragment: dequant one 32-wide quant block (== WMMA K) ----
        // B 32x16 layout: lane L -> n = L%16, kg = L/16; element e of the
        // v16h holds K = kg*16 + e. Lane reads 16 consecutive int32 codes
        // = one aligned 64B chunk of the 128B quant block.
        const int*  wp    = wrow + b * 32;
        const float scale = srow[b];
        int q[16];
        *(v4i*)(q + 0)  = *(const v4i*)(wp + 0);
        *(v4i*)(q + 4)  = *(const v4i*)(wp + 4);
        *(v4i*)(q + 8)  = *(const v4i*)(wp + 8);
        *(v4i*)(q + 12) = *(const v4i*)(wp + 12);
        v16h bf;
        #pragma unroll
        for (int i = 0; i < 16; ++i)
            bf[i] = (_Float16)(((float)(q[i] - 128)) * scale);

        // ---- A fragments: fp32 x -> f16, 4 M-tiles reuse this B ----
        // A 16x32 layout: lane L -> m = L%16, h = L/16;
        // elements 0..15 <- K in [h*16, h*16+16)
        // x is only 0.77 MB -> fully L2-resident, redundant reads are cheap.
        const int k0 = b * 32 + half * 16;
        const int k1 = k0 + 8;
        #pragma unroll
        for (int mt = 0; mt < 4; ++mt) {
            const float* xp = x + (size_t)(mt * 16 + nloc) * IN_F;
            v4f f0 = *(const v4f*)(xp + k0);
            v4f f1 = *(const v4f*)(xp + k0 + 4);
            v4f f2 = *(const v4f*)(xp + k1);
            v4f f3 = *(const v4f*)(xp + k1 + 4);
            v16h av;
            #pragma unroll
            for (int i = 0; i < 4; ++i) {
                av[i]      = (_Float16)f0[i];
                av[i + 4]  = (_Float16)f1[i];
                av[i + 8]  = (_Float16)f2[i];
                av[i + 12] = (_Float16)f3[i];
            }
            // (neg_a, A, neg_b, B, c_mod, C, reuse_a, reuse_b)
            acc[mt] = __builtin_amdgcn_wmma_f32_16x16x32_f16(
                false, av, false, bf, (short)0, acc[mt], false, false);
        }
    }

    // ---- K-split reduction through LDS (16 KB), then bias + store ----
    __shared__ float red[KSPLIT][4][8][32];
    #pragma unroll
    for (int mt = 0; mt < 4; ++mt)
        #pragma unroll
        for (int r = 0; r < 8; ++r)
            red[wv][mt][r][lane] = acc[mt][r];
    __syncthreads();

    // Wave wv reduces M-tile wv across the 4 K-split partials.
    // C layout: VGPR r, lane L -> m = r + 8*(L/16), n = L%16.
    const float bias = ((float)(bq[ncol] - 128)) * bsc[ncol >> 5];
    #pragma unroll
    for (int r = 0; r < 8; ++r) {
        float s = red[0][wv][r][lane] + red[1][wv][r][lane]
                + red[2][wv][r][lane] + red[3][wv][r][lane];
        const int m = wv * 16 + half * 8 + r;
        out[(size_t)m * OUT_F + ncol] = s + bias;
    }
}

extern "C" void kernel_launch(void* const* d_in, const int* in_sizes, int n_in,
                              void* d_out, int out_size, void* d_ws, size_t ws_size,
                              hipStream_t stream) {
    const float* x   = (const float*)d_in[0];
    const int*   wq  = (const int*)  d_in[1];
    const float* wsc = (const float*)d_in[2];
    const int*   bq  = (const int*)  d_in[3];
    const float* bsc = (const float*)d_in[4];
    float* out = (float*)d_out;

    dim3 grid(OUT_F / 16);        // 768 N-tiles
    dim3 block(32 * KSPLIT);      // 4 waves, K-split
    hipLaunchKernelGGL(DequantingLinear_52321291600113_kernel, grid, block, 0, stream,
                       x, wq, wsc, bq, bsc, out);

    (void)in_sizes; (void)n_in; (void)d_ws; (void)ws_size; (void)out_size;
}
